// JADirectParamGRU_79199196938502
// MI455X (gfx1250) — matrix operand order repaired and checked by reading, not compiled, not run on hardware
//
#include <hip/hip_runtime.h>

// ---------------------------------------------------------------------------
// Jiles-Atherton GRU scan, CDNA5 (gfx1250) wave32 WMMA implementation.
//
// Grid: 32 workgroups (one per 16-row batch tile), 128 threads = 4 waves.
// Wave w owns hidden columns [16w, 16w+16): computes r/z/inn/hn gate tiles
// with v_wmma_f32_16x16x32_f16, weights resident in VGPRs as f16 fragments.
//
// Latency-critical-path optimizations:
//  * double-buffered x/h LDS tiles -> exactly ONE s_barrier per timestep
//  * all divisions via v_rcp_f32 (TRANS, co-executes with WMMA) instead of
//    IEEE div_scale/div_fixup expansions
//  * features / B_seq software-pipelined one step ahead + global_prefetch
// ---------------------------------------------------------------------------

typedef __attribute__((ext_vector_type(16))) _Float16 v16h;
typedef __attribute__((ext_vector_type(8)))  _Float16 v8h;
typedef __attribute__((ext_vector_type(8)))  float    v8f;

#define TT     4096
#define FF     15
#define HD     64
#define BSZ    512
#define MTILE  16
#define NFE    (MTILE * FF)     // 240 feature elements per tile per step

#define MU0F     1.2566370614359172e-06f
#define INV_MU0  795774.715459477f

__device__ __forceinline__ float frcp(float x) {
  return __builtin_amdgcn_rcpf(x);              // v_rcp_f32 (fast)
}
__device__ __forceinline__ float sigm(float x) {
  return frcp(1.f + __expf(-x));
}
__device__ __forceinline__ float ftanh(float x) {
  x = fminf(fmaxf(x, -15.f), 15.f);
  const float e = __expf(2.f * x);
  return (e - 1.f) * frcp(e + 1.f);
}
__device__ __forceinline__ float fcoth(float x) {   // x != 0 guaranteed
  x = fminf(fmaxf(x, -15.f), 15.f);
  const float e = __expf(2.f * x);
  return (e + 1.f) * frcp(e - 1.f);
}
__device__ __forceinline__ float sgnf(float x) {
  return x > 0.f ? 1.f : (x < 0.f ? -1.f : 0.f);
}
__device__ __forceinline__ v8f wmma16(v16h a, v16h b, v8f c) {
  // D = A(16x32 f16) * B(32x16 f16) + C(16x16 f32)
  return __builtin_amdgcn_wmma_f32_16x16x32_f16(false, a, false, b,
                                                (short)0, c, false, false);
}

__global__ __launch_bounds__(128) void ja_gru_kernel(
    const float* __restrict__ H0,    const float* __restrict__ Bseq,
    const float* __restrict__ feat,  const float* __restrict__ h_init,
    const float* __restrict__ W_ih,  const float* __restrict__ W_hh,
    const float* __restrict__ b_ih,  const float* __restrict__ b_n,
    float* __restrict__ out)
{
  // double-buffered tiles: step t uses buffer t&1, writes buffer (t+1)&1
  __shared__ __align__(16) _Float16 xlds[2][MTILE * 16];
  __shared__ __align__(16) _Float16 hlds[2][MTILE * HD];
  __shared__ float pscr[MTILE * 5];     // wave-0 private (in-order DS)
  __shared__ float Hs[MTILE];           // wave-0 private

  const int tid  = threadIdx.x;
  const int lane = tid & 31;
  const int wv   = tid >> 5;       // wave 0..3 -> hidden cols 16*wv..+15
  const int grp  = lane >> 4;      // half-wave group (A/B K split)
  const int ln16 = lane & 15;
  const int b0   = blockIdx.x * MTILE;

  // ---- weight fragments: converted to f16 once, resident in VGPRs --------
  // B layout (16-bit, 32x16): element s<8 holds K=8*grp+s, s>=8 holds
  // K=16+8*grp+(s-8), for column n = lane%16.
  v16h Bi[3];          // gi fragments (K=16 real, upper K zero-padded)
  v16h Bh[3][2];       // gh fragments (two K=32 chunks of K=64)
  float bias_i[3];
  #pragma unroll
  for (int g = 0; g < 3; ++g) {
    const int grow = g * 64 + wv * 16 + ln16;   // output-gate row of W
    bias_i[g] = b_ih[grow];
    {
      const float* p = W_ih + grow * 16;
      #pragma unroll
      for (int i = 0; i < 16; ++i)
        Bi[g][i] = (i < 8) ? (_Float16)p[grp * 8 + i] : (_Float16)0.f;
    }
    #pragma unroll
    for (int kc = 0; kc < 2; ++kc) {
      const float* p = W_hh + grow * HD + kc * 32 + grp * 8;
      #pragma unroll
      for (int i = 0; i < 16; ++i)
        Bh[g][kc][i] = (_Float16)p[i + (i >= 8 ? 8 : 0)];
    }
  }
  const float bn_l = b_n[wv * 16 + ln16];

  // ---- hidden state: f32 fragment in C/D layout + f16 copy in LDS[0] -----
  const int hcol = wv * 16 + ln16;
  float hacc[8];
  #pragma unroll
  for (int v = 0; v < 8; ++v) {
    const int m = v + grp * 8;
    hacc[v] = h_init[(size_t)(b0 + m) * HD + hcol];
    hlds[0][m * HD + hcol] = (_Float16)hacc[v];
  }
  if (tid < MTILE) Hs[tid] = H0[b0 + tid];

  // ---- software pipeline: feature + B_seq prefetch registers -------------
  const int idx0 = tid, idx1 = tid + 128;
  const int r0 = idx0 / FF, c0 = idx0 % FF;
  const int r1 = idx1 / FF, c1 = idx1 % FF;
  float f0 = 0.f, f1 = 0.f;
  if (idx0 < NFE) f0 = feat[((size_t)(b0 + r0) * TT) * FF + c0];
  if (idx1 < NFE) f1 = feat[((size_t)(b0 + r1) * TT) * FF + c1];

  float Bp = 0.f, Bn = 0.f;
  if (wv == 0 && lane < MTILE) Bp = Bseq[(size_t)(b0 + lane) * (TT + 1)];

  for (int t = 0; t < TT; ++t) {
    const int p = t & 1;
    _Float16*       xb  = xlds[p];
    const _Float16* hb  = hlds[p];
    _Float16*       hbn = hlds[p ^ 1];

    // -- stage A: publish x = [H/H_MAX, feat_t] into buffer p ------------
    if (tid < MTILE) xb[tid * 16] = (_Float16)(Hs[tid] * (1.f / 100.f));
    if (idx0 < NFE) xb[r0 * 16 + 1 + c0] = (_Float16)f0;
    if (idx1 < NFE) xb[r1 * 16 + 1 + c1] = (_Float16)f1;
    __syncthreads();   // the ONLY barrier per timestep

    // -- stage B: issue next step's global loads early -------------------
    if (t + 1 < TT) {
      if (idx0 < NFE) f0 = feat[((size_t)(b0 + r0) * TT + (t + 1)) * FF + c0];
      if (idx1 < NFE) f1 = feat[((size_t)(b0 + r1) * TT + (t + 1)) * FF + c1];
      if (t + 2 < TT && idx0 < NFE)
        __builtin_prefetch(&feat[((size_t)(b0 + r0) * TT + (t + 2)) * FF + c0], 0, 0);
    }
    if (wv == 0 && lane < MTILE)
      Bn = Bseq[(size_t)(b0 + lane) * (TT + 1) + t + 1];

    // -- stage C: A fragments from LDS (16-bit A layout, b128 loads) -----
    const int m  = ln16;
    const int kb = grp * 8;
    v16h ax, ah0, ah1;
    {
      const v8h xlo = *reinterpret_cast<const v8h*>(&xb[m * 16 + kb]);
      const v8h h00 = *reinterpret_cast<const v8h*>(&hb[m * HD +  0 + kb]);
      const v8h h01 = *reinterpret_cast<const v8h*>(&hb[m * HD + 16 + kb]);
      const v8h h10 = *reinterpret_cast<const v8h*>(&hb[m * HD + 32 + kb]);
      const v8h h11 = *reinterpret_cast<const v8h*>(&hb[m * HD + 48 + kb]);
      #pragma unroll
      for (int i = 0; i < 8; ++i) {
        ax[i]  = xlo[i]; ax[i + 8]  = (_Float16)0.f;   // K=16..31 zero pad
        ah0[i] = h00[i]; ah0[i + 8] = h01[i];
        ah1[i] = h10[i]; ah1[i + 8] = h11[i];
      }
    }

    // -- stage D: 9 WMMAs (r/z fused input+hidden; inn/hn separate) ------
    v8f acc_r = {}, acc_z = {}, acc_in = {}, acc_hn = {};
    acc_r  = wmma16(ax,  Bi[0],    acc_r);
    acc_r  = wmma16(ah0, Bh[0][0], acc_r);
    acc_r  = wmma16(ah1, Bh[0][1], acc_r);
    acc_z  = wmma16(ax,  Bi[1],    acc_z);
    acc_z  = wmma16(ah0, Bh[1][0], acc_z);
    acc_z  = wmma16(ah1, Bh[1][1], acc_z);
    acc_in = wmma16(ax,  Bi[2],    acc_in);
    acc_hn = wmma16(ah0, Bh[2][0], acc_hn);
    acc_hn = wmma16(ah1, Bh[2][1], acc_hn);

    // -- stage E: GRU nonlinearity, f32 state update ---------------------
    float hn_v[8];
    #pragma unroll
    for (int v = 0; v < 8; ++v) {
      const float r = sigm(acc_r[v] + bias_i[0]);
      const float z = sigm(acc_z[v] + bias_i[1]);
      const float n = ftanh(acc_in[v] + bias_i[2] + r * (acc_hn[v] + bn_l));
      const float h = n + z * (hacc[v] - n);
      hacc[v] = h;
      hn_v[v] = h;
    }

    // -- stage F: publish new hidden (f16) into buffer p^1 ---------------
    #pragma unroll
    for (int v = 0; v < 8; ++v)
      hbn[(v + grp * 8) * HD + hcol] = (_Float16)hn_v[v];
    if (wv == 0 && ln16 < 5) {
      #pragma unroll
      for (int v = 0; v < 8; ++v)
        pscr[(v + grp * 8) * 5 + ln16] = hn_v[v];   // exact f32 p columns
    }

    // -- stage G: Jiles-Atherton physics (wave 0, one lane per row) ------
    // pscr/Hs are wave-0 private: LDS ops are in-order within a wave,
    // so no barrier is required between F and G or G and next A.
    if (wv == 0 && lane < MTILE) {
      const float p0 = pscr[lane * 5 + 0], p1 = pscr[lane * 5 + 1];
      const float p2 = pscr[lane * 5 + 2], p3 = pscr[lane * 5 + 3];
      const float p4 = pscr[lane * 5 + 4];
      const float Ms = 2.0e6f  * sigm(p0);
      const float a  = 100.f   * sigm(p1);
      const float al = 1.0e-4f * sigm(p2);
      const float k  = 100.f   * sigm(p3);
      const float c  = sigm(p4);
      const float dBdt  = (Bn - Bp) * 16.0e6f;          // (Bn-Bp)/TAU
      const float Hp    = Hs[lane];
      const float M     = Bp * INV_MU0 - Hp;
      const float He    = Hp + al * M;
      const float rcp_a = frcp(a);
      float xr = He * rcp_a;
      xr = (fabsf(xr) < 1e-7f) ? 1e-7f * sgnf(xr) : xr;
      const float coth = fcoth(xr);
      const float aHe  = a * frcp(He);
      const float Man  = Ms * (coth - aHe);
      const float dm   = 0.5f * (1.f + sgnf((Man - M) * dBdt));
      const float dMan = Ms * rcp_a * (1.f - coth * coth + aHe * aHe);
      const float dlt  = sgnf(dBdt);
      const float num  = dm * (Man - M) + c * k * dlt * dMan;
      const float den  = k * dlt - al * num;
      const float dMdH = num * frcp(den);
      const float dMdB = dMdH * frcp(MU0F * (1.f + dMdH));
      const float dMdt = dMdB * dBdt;
      const float dHdt = 1.0e8f * ftanh((dBdt * INV_MU0 - dMdt) * 1.0e-8f);
      const float Hn2  = Hp + dHdt * (1.f / 16.0e6f);   // TAU*dHdt
      Hs[lane] = Hn2;
      out[(size_t)(b0 + lane) * TT + t] = Hn2;
      Bp = Bn;
    }
  }
}

extern "C" void kernel_launch(void* const* d_in, const int* in_sizes, int n_in,
                              void* d_out, int out_size, void* d_ws, size_t ws_size,
                              hipStream_t stream) {
  (void)in_sizes; (void)n_in; (void)out_size; (void)d_ws; (void)ws_size;
  const float* H0   = (const float*)d_in[0];
  const float* Bseq = (const float*)d_in[1];
  const float* feat = (const float*)d_in[2];
  const float* hin  = (const float*)d_in[3];
  const float* Wih  = (const float*)d_in[4];
  const float* Whh  = (const float*)d_in[5];
  const float* bih  = (const float*)d_in[6];
  const float* bn   = (const float*)d_in[7];
  float* out = (float*)d_out;

  dim3 grid(BSZ / MTILE);   // 32 workgroups
  dim3 block(128);          // 4 wave32s
  ja_gru_kernel<<<grid, block, 0, stream>>>(H0, Bseq, feat, hin,
                                            Wih, Whh, bih, bn, out);
}
